// MaskedAutoencoderViM_21431886807214
// MI455X (gfx1250) — matrix operand order repaired and checked by reading, not compiled, not run on hardware
//
#include <hip/hip_runtime.h>
#include <hip/hip_bf16.h>
#include <math.h>

// ---------------- problem constants ----------------
#define PSZ   16
#define IMG   224
#define DMODEL 768
#define DEPTH 12
#define DI    1536
#define NST   16
#define DCONV 4
#define DTR   48
#define GS    14
#define LTOK  196
#define KEEP  49
#define BATCH 8
#define MV    (BATCH * KEEP)   // 392 valid token rows
#define MP    448              // padded rows (7 groups of 64)
#define KDT   64               // padded dt-projection K
#define EPSV  1e-5f

typedef __attribute__((ext_vector_type(16))) __bf16 bf16x16;
typedef __attribute__((ext_vector_type(8)))  __bf16 bf16x8;
typedef __attribute__((ext_vector_type(8)))  float  floatx8;

// ---------------------------------------------------------------------------
// WMMA bf16 GEMM:  C[M,N] (f32) = A[M,K] (bf16) * W[N,K]^T (bf16)
// Operands pre-converted to bf16 (halves L2/HBM bytes, no cvt in k-loop).
// A fragment (16x32): lane&15 = row M; lanes 0-15 hold K {0..7,16..23},
// lanes 16-31 hold K {8..15,24..31} (ISA 7.12.2 table).
// B fragment (32x16): lane&15 = col N; lanes 0-15 hold K {0..7,16..23}, lanes 16-31
// hold K {8..15,24..31}. C/D: lane&15 = col N; VGPR r = row r / 8+r.
// ---------------------------------------------------------------------------
__device__ __forceinline__ bf16x16 join8(bf16x8 lo, bf16x8 hi) {
  bf16x16 f;
#pragma unroll
  for (int i = 0; i < 8; ++i) { f[i] = lo[i]; f[8 + i] = hi[i]; }
  return f;
}

__device__ __forceinline__ bf16x16 frag_a(const __bf16* __restrict__ A, int lda,
                                          int row, int k0, int lane) {
  const __bf16* p = A + (size_t)row * lda + k0 + ((lane >> 4) << 3);   // +0 / +8
  return join8(*(const bf16x8*)p, *(const bf16x8*)(p + 16));
}

__device__ __forceinline__ bf16x16 frag_b(const __bf16* __restrict__ W, int ldb,
                                          int ncol, int k0, int lane) {
  const __bf16* p = W + (size_t)ncol * ldb + k0 + ((lane >> 4) << 3);  // +0 / +8
  return join8(*(const bf16x8*)p, *(const bf16x8*)(p + 16));
}

// EPI: 0 = f32 store (+opt bias), 1 = bias+softplus, 2 = xdbl split store
//      (EPI2: nc<48 -> bf16 dtr buffer [row,64]; nc>=48 -> f32 B/C buffer [row,32])
template <int EPI>
__global__ __launch_bounds__(128)
void gemm_wmma_kernel(const __bf16* __restrict__ A, const __bf16* __restrict__ W,
                      float* __restrict__ C, __bf16* __restrict__ Cb,
                      const float* __restrict__ bias,
                      int Mvalid, int K, int N, int lda, int ldc) {
  const int lane   = threadIdx.x & 31;
  const int wave   = threadIdx.x >> 5;
  const int ntiles = N >> 4;
  const int ntile  = blockIdx.x * 4 + wave;
  if (ntile >= ntiles) return;                      // wave-uniform exit
  const int n0   = ntile << 4;
  const int m0   = blockIdx.y << 6;                 // 64-row M group
  const int mrow = lane & 15;

  floatx8 acc0 = {}; floatx8 acc1 = {}; floatx8 acc2 = {}; floatx8 acc3 = {};

  for (int k0 = 0; k0 < K; k0 += 32) {
    bf16x16 bf = frag_b(W, K, n0 + mrow, k0, lane);
    bf16x16 a0 = frag_a(A, lda, m0 +  0 + mrow, k0, lane);
    bf16x16 a1 = frag_a(A, lda, m0 + 16 + mrow, k0, lane);
    bf16x16 a2 = frag_a(A, lda, m0 + 32 + mrow, k0, lane);
    bf16x16 a3 = frag_a(A, lda, m0 + 48 + mrow, k0, lane);
    acc0 = __builtin_amdgcn_wmma_f32_16x16x32_bf16(false, a0, false, bf, (short)0, acc0, false, false);
    acc1 = __builtin_amdgcn_wmma_f32_16x16x32_bf16(false, a1, false, bf, (short)0, acc1, false, false);
    acc2 = __builtin_amdgcn_wmma_f32_16x16x32_bf16(false, a2, false, bf, (short)0, acc2, false, false);
    acc3 = __builtin_amdgcn_wmma_f32_16x16x32_bf16(false, a3, false, bf, (short)0, acc3, false, false);
  }

  const int nc = n0 + mrow;
  const float bv = (EPI == 1) ? bias[nc] : ((EPI == 0 && bias) ? bias[nc] : 0.0f);
  const int mb = m0 + ((lane >> 4) << 3);

  auto store8 = [&](const floatx8& acc, int mbase) {
#pragma unroll
    for (int r = 0; r < 8; ++r) {
      int row = mbase + r;
      if (row < Mvalid) {
        float v = acc[r] + bv;
        if (EPI == 1) v = (v > 20.0f) ? v : log1pf(expf(v));   // softplus
        if (EPI == 2) {
          if (nc < DTR) Cb[(size_t)row * KDT + nc] = (__bf16)v;  // dtr (bf16)
          else          C[(size_t)row * 32 + (nc - DTR)] = v;    // B/C (f32)
        } else {
          C[(size_t)row * ldc + nc] = v;
        }
      }
    }
  };
  store8(acc0, mb); store8(acc1, mb + 16); store8(acc2, mb + 32); store8(acc3, mb + 48);
}

// ---------------------------------------------------------------------------
// One-time weight preparation
__global__ __launch_bounds__(256)
void f32_to_bf16_kernel(const float* __restrict__ src, __bf16* __restrict__ dst, int n) {
  const int i = blockIdx.x * 256 + threadIdx.x;
  if (i < n) dst[i] = (__bf16)src[i];
}

// Zero-pad dt_w (12,1536,48) -> bf16 (12,1536,64) so the dt GEMM has K%32==0.
__global__ __launch_bounds__(256)
void pad_dtw_kernel(const float* __restrict__ dtw, __bf16* __restrict__ dtp) {
  const int idx = blockIdx.x * 256 + threadIdx.x;
  if (idx >= DEPTH * DI * KDT) return;
  const int col = idx & 63, r = idx >> 6;     // r = layer*1536 + row
  dtp[idx] = (col < DTR) ? (__bf16)dtw[(size_t)r * DTR + col] : (__bf16)0.0f;
}

// Zero the pad columns 48..63 of the bf16 dtr buffer (never rewritten after).
__global__ __launch_bounds__(256)
void zero_dtr_pad_kernel(__bf16* __restrict__ dtr) {
  const int idx = blockIdx.x * 256 + threadIdx.x;
  if (idx >= MP * 16) return;
  const int rrow = idx >> 4, c = idx & 15;
  dtr[(size_t)rrow * KDT + DTR + c] = (__bf16)0.0f;
}

// ---------------------------------------------------------------------------
// Token masking: stable rank-sort of noise per batch (O(196^2), trivial).
__global__ __launch_bounds__(256)
void mask_sort_kernel(const float* __restrict__ noise, float* __restrict__ mask_out,
                      int* __restrict__ ids_restore_out, int* __restrict__ ids_keep) {
  __shared__ float nv[LTOK];
  const int b = blockIdx.x, tid = threadIdx.x;
  if (tid < LTOK) nv[tid] = noise[b * LTOK + tid];
  __syncthreads();
  if (tid < LTOK) {
    const float vi = nv[tid];
    int r = 0;
    for (int j = 0; j < LTOK; ++j) {
      float vj = nv[j];
      r += (vj < vi) || (vj == vi && j < tid);     // stable (matches argsort)
    }
    mask_out[b * LTOK + tid]        = (r < KEEP) ? 0.0f : 1.0f;
    ids_restore_out[b * LTOK + tid] = r;
    if (r < KEEP) ids_keep[b * KEEP + r] = tid;
  }
}

// Gather kept patches into a bf16 [MP, 768] A-matrix (row t = token b*49+l).
__global__ __launch_bounds__(256)
void gather_patch_kernel(const float* __restrict__ imgs, const int* __restrict__ ids_keep,
                         __bf16* __restrict__ Ap) {
  const int t = blockIdx.x;                 // 0..391
  const int b = t / KEEP;
  const int pid = ids_keep[t];
  const int gy = pid / GS, gx = pid % GS;
  const float* base = imgs + (size_t)b * 3 * IMG * IMG;
  for (int k = threadIdx.x; k < DMODEL; k += blockDim.x) {
    const int c = k >> 8, rem = k & 255, i = rem >> 4, j = rem & 15;
    Ap[(size_t)t * DMODEL + k] =
        (__bf16)base[((size_t)c * IMG + gy * PSZ + i) * IMG + gx * PSZ + j];
  }
}

// hs += pos_embed[kept patch]; res = 0
__global__ __launch_bounds__(256)
void add_pos_init_kernel(float* __restrict__ hs, const float* __restrict__ pos,
                         const int* __restrict__ ids_keep, float* __restrict__ res) {
  const int t = blockIdx.x;
  const int pid = ids_keep[t];
  for (int d = threadIdx.x; d < DMODEL; d += blockDim.x) {
    hs[(size_t)t * DMODEL + d] += pos[(size_t)pid * DMODEL + d];
    res[(size_t)t * DMODEL + d] = 0.0f;
  }
}

__device__ __forceinline__ void block_reduce2(float& s1, float& s2) {
  __shared__ float sA[256], sB[256];
  sA[threadIdx.x] = s1; sB[threadIdx.x] = s2;
  __syncthreads();
  for (int s = 128; s > 0; s >>= 1) {
    if ((int)threadIdx.x < s) { sA[threadIdx.x] += sA[threadIdx.x + s];
                                sB[threadIdx.x] += sB[threadIdx.x + s]; }
    __syncthreads();
  }
  s1 = sA[0]; s2 = sB[0];
  __syncthreads();
}

// res += hs (stored back); out(bf16) = LN(res) over DMODEL  (GEMM-A only)
__global__ __launch_bounds__(256)
void add_ln_kernel(float* __restrict__ res, const float* __restrict__ hs,
                   __bf16* __restrict__ out, const float* __restrict__ w,
                   const float* __restrict__ bq) {
  const int t = blockIdx.x;
  float s1 = 0.0f, s2 = 0.0f;
  for (int d = threadIdx.x; d < DMODEL; d += 256) {
    float v = res[(size_t)t * DMODEL + d] + hs[(size_t)t * DMODEL + d];
    res[(size_t)t * DMODEL + d] = v;
    s1 += v; s2 += v * v;
  }
  block_reduce2(s1, s2);
  const float mu = s1 / DMODEL;
  const float rs = rsqrtf(s2 / DMODEL - mu * mu + EPSV);
  for (int d = threadIdx.x; d < DMODEL; d += 256)
    out[(size_t)t * DMODEL + d] =
        (__bf16)((res[(size_t)t * DMODEL + d] - mu) * rs * w[d] + bq[d]);
}

// depthwise causal conv (K=4) + SiLU; writes xc in f32 (scan) and bf16 (GEMM A)
__global__ __launch_bounds__(256)
void conv_silu_kernel(const float* __restrict__ xz, const float* __restrict__ cw,
                      const float* __restrict__ cb, float* __restrict__ xc,
                      __bf16* __restrict__ xcb) {
  const int idx = blockIdx.x * blockDim.x + threadIdx.x;
  if (idx >= MV * DI) return;
  const int t = idx / DI, c = idx % DI, l = t % KEEP;
  float acc = cb[c];
#pragma unroll
  for (int k = 0; k < DCONV; ++k) {
    const int ll = l - (DCONV - 1) + k;
    if (ll >= 0)
      acc += xz[(size_t)(t - (DCONV - 1) + k) * (2 * DI) + c] * cw[c * DCONV + k];
  }
  const float s = acc / (1.0f + __expf(-acc));    // SiLU
  xc[idx]  = s;
  xcb[idx] = (__bf16)s;
}

// selective scan: one thread per (batch, channel); 16 states in registers.
// bc layout: [MP, 32] f32, B at cols 0..15, C at cols 16..31.
__global__ __launch_bounds__(256)
void scan_kernel(const float* __restrict__ dt, const float* __restrict__ xc,
                 const float* __restrict__ bc, const float* __restrict__ Alog,
                 const float* __restrict__ Dsk, float* __restrict__ y) {
  const int idx = blockIdx.x * blockDim.x + threadIdx.x;
  if (idx >= BATCH * DI) return;
  const int b = idx / DI, d = idx % DI;
  float a[NST], st[NST];
#pragma unroll
  for (int n = 0; n < NST; ++n) { a[n] = -__expf(Alog[d * NST + n]); st[n] = 0.0f; }
  const float dsk = Dsk[d];
  for (int l = 0; l < KEEP; ++l) {
    const int t = b * KEEP + l;
    const float dtv = dt[(size_t)t * DI + d];
    const float xcv = xc[(size_t)t * DI + d];
    const float dx  = dtv * xcv;
    const float* row = bc + (size_t)t * 32;
    float acc = 0.0f;
#pragma unroll
    for (int n = 0; n < NST; ++n) {
      const float dA = __expf(dtv * a[n]);
      st[n] = dA * st[n] + dx * row[n];                // B block
      acc  += st[n] * row[16 + n];                     // C block
    }
    y[(size_t)t * DI + d] = acc + dsk * xcv;
  }
}

// out(bf16) = LN(y) * silu(z)    (z = second half of xz row; GEMM-A only)
__global__ __launch_bounds__(256)
void ssm_ln_gate_kernel(const float* __restrict__ y, const float* __restrict__ xz,
                        const float* __restrict__ w, const float* __restrict__ bq,
                        __bf16* __restrict__ out) {
  const int t = blockIdx.x;
  float s1 = 0.0f, s2 = 0.0f;
  for (int c = threadIdx.x; c < DI; c += 256) {
    float v = y[(size_t)t * DI + c];
    s1 += v; s2 += v * v;
  }
  block_reduce2(s1, s2);
  const float mu = s1 / DI;
  const float rs = rsqrtf(s2 / DI - mu * mu + EPSV);
  for (int c = threadIdx.x; c < DI; c += 256) {
    float v = (y[(size_t)t * DI + c] - mu) * rs * w[c] + bq[c];
    float z = xz[(size_t)t * 2 * DI + DI + c];
    out[(size_t)t * DI + c] = (__bf16)(v * (z / (1.0f + __expf(-z))));
  }
}

// latent = LN(res + hs) written straight to d_out
__global__ __launch_bounds__(256)
void final_ln_kernel(const float* __restrict__ res, const float* __restrict__ hs,
                     const float* __restrict__ w, const float* __restrict__ bq,
                     float* __restrict__ out) {
  const int t = blockIdx.x;
  __shared__ float rowv[DMODEL];
  float s1 = 0.0f, s2 = 0.0f;
  for (int d = threadIdx.x; d < DMODEL; d += 256) {
    float v = res[(size_t)t * DMODEL + d] + hs[(size_t)t * DMODEL + d];
    rowv[d] = v; s1 += v; s2 += v * v;
  }
  block_reduce2(s1, s2);
  const float mu = s1 / DMODEL;
  const float rs = rsqrtf(s2 / DMODEL - mu * mu + EPSV);
  for (int d = threadIdx.x; d < DMODEL; d += 256)
    out[(size_t)t * DMODEL + d] = (rowv[d] - mu) * rs * w[d] + bq[d];
}

// ---------------------------------------------------------------------------
extern "C" void kernel_launch(void* const* d_in, const int* in_sizes, int n_in,
                              void* d_out, int out_size, void* d_ws, size_t ws_size,
                              hipStream_t stream) {
  (void)in_sizes; (void)n_in; (void)out_size; (void)ws_size;
  const float* imgs      = (const float*)d_in[0];
  const float* noise     = (const float*)d_in[1];
  const float* patch_w   = (const float*)d_in[2];   // (768, 768) row-major over (3,16,16)
  const float* patch_b   = (const float*)d_in[3];
  const float* pos_embed = (const float*)d_in[4];
  const float* ln_w      = (const float*)d_in[5];
  const float* ln_b      = (const float*)d_in[6];
  const float* in_proj_w = (const float*)d_in[7];   // (12, 3072, 768)
  const float* conv_w    = (const float*)d_in[8];   // (12, 1536, 1, 4)
  const float* conv_b    = (const float*)d_in[9];
  const float* x_proj_w  = (const float*)d_in[10];  // (12, 80, 1536)
  const float* dt_w      = (const float*)d_in[11];  // (12, 1536, 48)
  const float* dt_b      = (const float*)d_in[12];
  const float* A_log     = (const float*)d_in[13];  // (12, 1536, 16)
  const float* D_skip    = (const float*)d_in[14];
  const float* ssm_ln_w  = (const float*)d_in[15];
  const float* ssm_ln_b  = (const float*)d_in[16];
  const float* out_proj_w= (const float*)d_in[17];  // (12, 768, 1536)
  const float* final_w   = (const float*)d_in[18];
  const float* final_b   = (const float*)d_in[19];

  float* out         = (float*)d_out;
  float* mask_out    = out + (size_t)MV * DMODEL;                  // 301056
  int*   ids_restore = (int*)(out + (size_t)MV * DMODEL + BATCH * LTOK);

  // ---- workspace layout (f32 region, then bf16 region; all 32B-aligned) ----
  float* ws = (float*)d_ws;
  size_t off = 0;
  auto falloc = [&](size_t n) { float* p = ws + off; off += n; return p; };
  float* hs    = falloc((size_t)MP * DMODEL);
  float* res   = falloc((size_t)MP * DMODEL);
  float* xz    = falloc((size_t)MP * 2 * DI);
  float* xc    = falloc((size_t)MP * DI);
  float* bc    = falloc((size_t)MP * 32);         // f32 B/C for the scan
  float* dtbuf = falloc((size_t)MP * DI);
  float* ybuf  = falloc((size_t)MP * DI);
  int*   ids_keep = (int*)(ws + off); off += MV + 8;   // keep 32B alignment

  __bf16* bws = (__bf16*)(ws + off);
  size_t boff = 0;
  auto balloc = [&](size_t n) { __bf16* p = bws + boff; boff += n; return p; };
  __bf16* Ap      = balloc((size_t)MP * DMODEL);
  __bf16* hln     = balloc((size_t)MP * DMODEL);
  __bf16* xcb     = balloc((size_t)MP * DI);
  __bf16* dtrb    = balloc((size_t)MP * KDT);      // bf16 dtr (cols 48..63 zero)
  __bf16* y2      = balloc((size_t)MP * DI);
  __bf16* patchwb = balloc((size_t)DMODEL * DMODEL);
  __bf16* ipwb    = balloc((size_t)DEPTH * 2 * DI * DMODEL);
  __bf16* xpwb    = balloc((size_t)DEPTH * 80 * DI);
  __bf16* dtwb    = balloc((size_t)DEPTH * DI * KDT);
  __bf16* opwb    = balloc((size_t)DEPTH * DMODEL * DI);

  auto launch_gemm = [&](const __bf16* A, const __bf16* W, float* Cc, __bf16* Cb,
                         const float* bias, int K, int N, int lda, int ldc, int epi) {
    dim3 grid((N / 16 + 3) / 4, MP / 64);
    if (epi == 1)      gemm_wmma_kernel<1><<<grid, 128, 0, stream>>>(A, W, Cc, Cb, bias, MV, K, N, lda, ldc);
    else if (epi == 2) gemm_wmma_kernel<2><<<grid, 128, 0, stream>>>(A, W, Cc, Cb, bias, MV, K, N, lda, ldc);
    else               gemm_wmma_kernel<0><<<grid, 128, 0, stream>>>(A, W, Cc, Cb, bias, MV, K, N, lda, ldc);
  };
  auto cvt = [&](const float* s, __bf16* d, size_t n) {
    f32_to_bf16_kernel<<<(unsigned)((n + 255) / 256), 256, 0, stream>>>(s, d, (int)n);
  };

  // ---- one-time prep: masking, bf16 weights, pads, patch embed ----
  mask_sort_kernel<<<BATCH, 256, 0, stream>>>(noise, mask_out, ids_restore, ids_keep);
  cvt(patch_w,    patchwb, (size_t)DMODEL * DMODEL);
  cvt(in_proj_w,  ipwb,    (size_t)DEPTH * 2 * DI * DMODEL);
  cvt(x_proj_w,   xpwb,    (size_t)DEPTH * 80 * DI);
  cvt(out_proj_w, opwb,    (size_t)DEPTH * DMODEL * DI);
  pad_dtw_kernel<<<(DEPTH * DI * KDT + 255) / 256, 256, 0, stream>>>(dt_w, dtwb);
  zero_dtr_pad_kernel<<<(MP * 16 + 255) / 256, 256, 0, stream>>>(dtrb);
  gather_patch_kernel<<<MV, 256, 0, stream>>>(imgs, ids_keep, Ap);
  launch_gemm(Ap, patchwb, hs, nullptr, patch_b, DMODEL, DMODEL, DMODEL, DMODEL, 0);
  add_pos_init_kernel<<<MV, 256, 0, stream>>>(hs, pos_embed, ids_keep, res);

  // ---- 12 Mamba blocks ----
  for (int Lr = 0; Lr < DEPTH; ++Lr) {
    const float*  lnw  = ln_w     + (size_t)Lr * DMODEL;
    const float*  lnb  = ln_b     + (size_t)Lr * DMODEL;
    const __bf16* ipw  = ipwb     + (size_t)Lr * 2 * DI * DMODEL;
    const float*  cw   = conv_w   + (size_t)Lr * DI * DCONV;
    const float*  cb   = conv_b   + (size_t)Lr * DI;
    const __bf16* xpw  = xpwb     + (size_t)Lr * 80 * DI;
    const __bf16* dtwp = dtwb     + (size_t)Lr * DI * KDT;
    const float*  dtbs = dt_b     + (size_t)Lr * DI;
    const float*  alog = A_log    + (size_t)Lr * DI * NST;
    const float*  dsk  = D_skip   + (size_t)Lr * DI;
    const float*  slw  = ssm_ln_w + (size_t)Lr * DI;
    const float*  slb  = ssm_ln_b + (size_t)Lr * DI;
    const __bf16* opw  = opwb     + (size_t)Lr * DMODEL * DI;

    add_ln_kernel<<<MV, 256, 0, stream>>>(res, hs, hln, lnw, lnb);
    launch_gemm(hln, ipw, xz, nullptr, nullptr, DMODEL, 2 * DI, DMODEL, 2 * DI, 0);
    conv_silu_kernel<<<(MV * DI + 255) / 256, 256, 0, stream>>>(xz, cw, cb, xc, xcb);
    launch_gemm(xcb, xpw, bc, dtrb, nullptr, DI, 80, DI, 32, 2);       // split epilogue
    launch_gemm(dtrb, dtwp, dtbuf, nullptr, dtbs, KDT, DI, KDT, DI, 1);// softplus epilogue
    scan_kernel<<<(BATCH * DI + 255) / 256, 256, 0, stream>>>(dtbuf, xc, bc, alog, dsk, ybuf);
    ssm_ln_gate_kernel<<<MV, 256, 0, stream>>>(ybuf, xz, slw, slb, y2);
    launch_gemm(y2, opw, hs, nullptr, nullptr, DI, DMODEL, DI, DMODEL, 0);
  }

  // ---- final LN -> latent in d_out ----
  final_ln_kernel<<<MV, 256, 0, stream>>>(res, hs, final_w, final_b, out);
}
